// RWKV7Encoder_57509612093801
// MI455X (gfx1250) — compile-verified, matrix operand order checked by reading, not run on hardware
//
#include <hip/hip_runtime.h>
#include <hip/hip_bf16.h>
#include <stdint.h>

// ---------------------------------------------------------------------------
// Types for CDNA5 WMMA (wave32, 16x16x32 bf16 -> f32)
// ---------------------------------------------------------------------------
typedef __bf16 bf16_t;
typedef bf16_t v16bf __attribute__((ext_vector_type(16)));
typedef bf16_t v8bf  __attribute__((ext_vector_type(8)));
typedef float  v8f   __attribute__((ext_vector_type(8)));

static __device__ __forceinline__ unsigned short f2bf(float f) {
    union { float f; uint32_t u; } x; x.f = f;
    uint32_t u = x.u;
    return (unsigned short)((u + 0x7FFFu + ((u >> 16) & 1u)) >> 16);
}
static __device__ __forceinline__ float sigm(float x) {
    return 1.0f / (1.0f + expf(-x));
}

// Load one 16-bf16 fragment (two contiguous 16B chunks per lane, matching the
// documented 16-bit A/B VGPR layout: K = kb + 16*(e>>3) + 8*half + (e&7)).
static __device__ __forceinline__ v16bf ldfrag(const unsigned short* __restrict__ p,
                                               int o0, int o1) {
    v8bf lo = *(const v8bf*)(const void*)(p + o0);
    v8bf hi = *(const v8bf*)(const void*)(p + o1);
    return __builtin_shufflevector(lo, hi, 0,1,2,3,4,5,6,7,8,9,10,11,12,13,14,15);
}
static __device__ __forceinline__ v8f wmma_bf16(v16bf a, v16bf b, v8f c) {
    return __builtin_amdgcn_wmma_f32_16x16x32_bf16(false, a, false, b,
                                                   (short)0, c, false, false);
}

// ---------------------------------------------------------------------------
// Main GEMM: C[M,N](f32) = A[M,K](bf16 rowmajor) x B (Bt[N,K] bf16 rowmajor)
// block = 256 threads (8 waves); wave -> 16x64 tile; block tile 128x64.
// Software-pipelined: fragments for K-block k+1 load while block k computes.
// grid = (N/64, M/128). Requires M%128==0, N%64==0, K%32==0.
// ---------------------------------------------------------------------------
__global__ void __launch_bounds__(256)
gemm_bf16_wmma64(const unsigned short* __restrict__ A,
                 const unsigned short* __restrict__ Bt,
                 float* __restrict__ C, int M, int N, int K)
{
    const int wave = threadIdx.x >> 5;
    const int lane = threadIdx.x & 31;
    const int half = lane >> 4;
    const int l15  = lane & 15;
    const int row0 = blockIdx.y * 128 + wave * 16;
    const int col0 = blockIdx.x * 64;

    const unsigned short* aRow = A  + (size_t)(row0 + l15) * K;
    const unsigned short* bR0  = Bt + (size_t)(col0 +  0 + l15) * K;
    const unsigned short* bR1  = Bt + (size_t)(col0 + 16 + l15) * K;
    const unsigned short* bR2  = Bt + (size_t)(col0 + 32 + l15) * K;
    const unsigned short* bR3  = Bt + (size_t)(col0 + 48 + l15) * K;

    v8f acc0 = {}, acc1 = {}, acc2 = {}, acc3 = {};

    // prologue: fragments for K-block 0
    int o0 = 8 * half, o1 = o0 + 16;
    v16bf aC  = ldfrag(aRow, o0, o1);
    v16bf b0C = ldfrag(bR0,  o0, o1);
    v16bf b1C = ldfrag(bR1,  o0, o1);
    v16bf b2C = ldfrag(bR2,  o0, o1);
    v16bf b3C = ldfrag(bR3,  o0, o1);

    for (int kb = 32; kb < K; kb += 32) {
        const int n0 = kb + 8 * half, n1 = n0 + 16;
        // issue next block's loads first ...
        v16bf aN  = ldfrag(aRow, n0, n1);
        v16bf b0N = ldfrag(bR0,  n0, n1);
        v16bf b1N = ldfrag(bR1,  n0, n1);
        v16bf b2N = ldfrag(bR2,  n0, n1);
        v16bf b3N = ldfrag(bR3,  n0, n1);
        // ... then compute on the block loaded one iteration ago
        acc0 = wmma_bf16(aC, b0C, acc0);
        acc1 = wmma_bf16(aC, b1C, acc1);
        acc2 = wmma_bf16(aC, b2C, acc2);
        acc3 = wmma_bf16(aC, b3C, acc3);
        aC = aN; b0C = b0N; b1C = b1N; b2C = b2N; b3C = b3N;
    }
    acc0 = wmma_bf16(aC, b0C, acc0);
    acc1 = wmma_bf16(aC, b1C, acc1);
    acc2 = wmma_bf16(aC, b2C, acc2);
    acc3 = wmma_bf16(aC, b3C, acc3);

    #pragma unroll
    for (int r = 0; r < 8; ++r) {
        int row = row0 + r + 8 * half;
        float* cp = C + (size_t)row * N + col0 + l15;
        cp[0]  = acc0[r];
        cp[16] = acc1[r];
        cp[32] = acc2[r];
        cp[48] = acc3[r];
    }
}

// ---------------------------------------------------------------------------
// Narrow GEMM (N%32==0 only; used for the Dv=32 projection).
// wave -> 16x32 tile; grid = (N/32, M/128).
// ---------------------------------------------------------------------------
__global__ void __launch_bounds__(256)
gemm_bf16_wmma32(const unsigned short* __restrict__ A,
                 const unsigned short* __restrict__ Bt,
                 float* __restrict__ C, int M, int N, int K)
{
    const int wave = threadIdx.x >> 5;
    const int lane = threadIdx.x & 31;
    const int half = lane >> 4;
    const int l15  = lane & 15;
    const int row0 = blockIdx.y * 128 + wave * 16;
    const int col0 = blockIdx.x * 32;

    const unsigned short* aRow  = A  + (size_t)(row0 + l15) * K;
    const unsigned short* bRow0 = Bt + (size_t)(col0 + l15) * K;
    const unsigned short* bRow1 = Bt + (size_t)(col0 + 16 + l15) * K;

    v8f acc0 = {}; v8f acc1 = {};
    for (int kb = 0; kb < K; kb += 32) {
        const int o0 = kb + 8 * half;
        const int o1 = o0 + 16;
        if (kb + 32 < K) {
            __builtin_prefetch(aRow  + o0 + 32, 0, 3);
            __builtin_prefetch(bRow0 + o0 + 32, 0, 3);
            __builtin_prefetch(bRow1 + o0 + 32, 0, 3);
        }
        v16bf af = ldfrag(aRow,  o0, o1);
        v16bf b0 = ldfrag(bRow0, o0, o1);
        v16bf b1 = ldfrag(bRow1, o0, o1);
        acc0 = wmma_bf16(af, b0, acc0);
        acc1 = wmma_bf16(af, b1, acc1);
    }
    #pragma unroll
    for (int r = 0; r < 8; ++r) {
        int row = row0 + r + 8 * half;
        C[(size_t)row * N + col0 + l15]      = acc0[r];
        C[(size_t)row * N + col0 + 16 + l15] = acc1[r];
    }
}

// ---------------------------------------------------------------------------
// Weight conversion: fp32 [L,K,N] -> bf16 transposed [L,N,K]
// ---------------------------------------------------------------------------
__global__ void k_transpose_bf16(const float* __restrict__ W,
                                 unsigned short* __restrict__ WT,
                                 int K, int N, long total)
{
    long idx = (long)blockIdx.x * 256 + threadIdx.x;
    if (idx >= total) return;
    long kn = (long)K * N;
    long l = idx / kn; long rem = idx - l * kn;
    int k = (int)(rem / N); int n = (int)(rem - (long)k * N);
    WT[l * kn + (long)n * K + k] = f2bf(W[idx]);
}

// ---------------------------------------------------------------------------
// LayerNorm over C, one block per token row
// ---------------------------------------------------------------------------
__global__ void __launch_bounds__(256)
k_layernorm(const float* __restrict__ x, const float* __restrict__ w,
            const float* __restrict__ b, float* __restrict__ out, int C)
{
    __shared__ float s[256];
    const int m = blockIdx.x, tid = threadIdx.x;
    const float* row = x + (size_t)m * C;
    float sum = 0.f;
    for (int c = tid; c < C; c += 256) sum += row[c];
    s[tid] = sum; __syncthreads();
    for (int o = 128; o > 0; o >>= 1) { if (tid < o) s[tid] += s[tid + o]; __syncthreads(); }
    float mean = s[0] / (float)C; __syncthreads();
    float vs = 0.f;
    for (int c = tid; c < C; c += 256) { float d = row[c] - mean; vs += d * d; }
    s[tid] = vs; __syncthreads();
    for (int o = 128; o > 0; o >>= 1) { if (tid < o) s[tid] += s[tid + o]; __syncthreads(); }
    float rstd = rsqrtf(s[0] / (float)C + 1e-5f);
    float* orow = out + (size_t)m * C;
    for (int c = tid; c < C; c += 256) orow[c] = (row[c] - mean) * rstd * w[c] + b[c];
}

// ---------------------------------------------------------------------------
// Token-shift + 6-way mix -> bf16
// ---------------------------------------------------------------------------
__global__ void k_mix6(const float* __restrict__ ln,
                       const float* __restrict__ xr, const float* __restrict__ xw,
                       const float* __restrict__ xk, const float* __restrict__ xv,
                       const float* __restrict__ xa, const float* __restrict__ xg,
                       unsigned short* mr, unsigned short* mw, unsigned short* mk,
                       unsigned short* mv, unsigned short* ma, unsigned short* mg,
                       int T, int C, int total)
{
    int idx = blockIdx.x * 256 + threadIdx.x;
    if (idx >= total) return;
    int c = idx % C; int m = idx / C; int t = m % T;
    float cur  = ln[idx];
    float prev = (t == 0) ? 0.0f : ln[idx - C];
    float xx = prev - cur;
    mr[idx] = f2bf(cur + xx * xr[c]);
    mw[idx] = f2bf(cur + xx * xw[c]);
    mk[idx] = f2bf(cur + xx * xk[c]);
    mv[idx] = f2bf(cur + xx * xv[c]);
    ma[idx] = f2bf(cur + xx * xa[c]);
    mg[idx] = f2bf(cur + xx * xg[c]);
}

// mode: 0 none, 1 tanh, 2 sigmoid, 3 relu^2
__global__ void k_convert(const float* __restrict__ in, unsigned short* __restrict__ out,
                          int total, int mode)
{
    int idx = blockIdx.x * 256 + threadIdx.x;
    if (idx >= total) return;
    float v = in[idx];
    if      (mode == 1) v = tanhf(v);
    else if (mode == 2) v = sigm(v);
    else if (mode == 3) { v = fmaxf(v, 0.f); v = v * v; }
    out[idx] = f2bf(v);
}

// decay = exp(-exp(-softplus(-z)-0.5)) = exp(-exp(-0.5)*sigmoid(z)), z = w0 + t
__global__ void k_decay(const float* __restrict__ z, const float* __restrict__ w0,
                        float* __restrict__ dec, int C, int total)
{
    int idx = blockIdx.x * 256 + threadIdx.x;
    if (idx >= total) return;
    int c = idx % C;
    dec[idx] = expf(-0.60653066f * sigm(w0[c] + z[idx]));
}

__global__ void k_asig(const float* __restrict__ z, const float* __restrict__ a0,
                       float* __restrict__ a, int C, int total)
{
    int idx = blockIdx.x * 256 + threadIdx.x;
    if (idx >= total) return;
    int c = idx % C;
    a[idx] = sigm(a0[c] + z[idx]);
}

__global__ void k_vmix(float* __restrict__ v, const float* __restrict__ vfirst,
                       const float* __restrict__ z, const float* __restrict__ v0,
                       int C, int total)
{
    int idx = blockIdx.x * 256 + threadIdx.x;
    if (idx >= total) return;
    int c = idx % C;
    float vv = v[idx];
    v[idx] = vv + (vfirst[idx] - vv) * sigm(v0[c] + z[idx]);
}

__global__ void k_add(float* __restrict__ x, const float* __restrict__ d, int total)
{
    int idx = blockIdx.x * 256 + threadIdx.x;
    if (idx >= total) return;
    x[idx] += d[idx];
}

__global__ void k_cmixin(const float* __restrict__ ln, const float* __restrict__ cxk,
                         unsigned short* __restrict__ out, int T, int C, int total)
{
    int idx = blockIdx.x * 256 + threadIdx.x;
    if (idx >= total) return;
    int c = idx % C; int m = idx / C; int t = m % T;
    float cur  = ln[idx];
    float prev = (t == 0) ? 0.0f : ln[idx - C];
    out[idx] = f2bf(cur + (prev - cur) * cxk[c]);
}

// ---------------------------------------------------------------------------
// kk normalization per head + b = kk*a + k' = k*(1+(a-1)*k_a)   (block = 64)
// ---------------------------------------------------------------------------
__global__ void __launch_bounds__(64)
k_kkprep(float* __restrict__ kio, const float* __restrict__ a,
         const float* __restrict__ kkv, const float* __restrict__ kav,
         float* __restrict__ kkn, float* __restrict__ bb, int H, int C)
{
    __shared__ float s[64];
    int bx = blockIdx.x; int h = bx % H; int m = bx / H;
    int tid = threadIdx.x;
    int c = h * 64 + tid;
    size_t idx = (size_t)m * C + c;
    float kv = kio[idx];
    float kk = kv * kkv[c];
    s[tid] = kk * kk; __syncthreads();
    for (int o = 32; o > 0; o >>= 1) { if (tid < o) s[tid] += s[tid + o]; __syncthreads(); }
    float inv = 1.0f / fmaxf(sqrtf(s[0]), 1e-12f);
    float kn = kk * inv;
    float av = a[idx];
    kkn[idx] = kn;
    bb[idx]  = kn * av;
    kio[idx] = kv * (1.0f + (av - 1.0f) * kav[c]);
}

// ---------------------------------------------------------------------------
// wkv7 sequential scan. One block per (b,h); 256 threads; 64x64 state in regs
// (thread t: row i = t&63, owns columns [16*(t>>6), +16)).
// ---------------------------------------------------------------------------
__global__ void __launch_bounds__(256)
k_wkv(const float* __restrict__ r, const float* __restrict__ wd,
      const float* __restrict__ k, const float* __restrict__ v,
      const float* __restrict__ kkn, const float* __restrict__ bb,
      float* __restrict__ y, int T, int C, int H)
{
    __shared__ float vec[384];   // r[0..63] w[64..] k[128..] v[192..] a[256..] b[320..]
    __shared__ float psum[256];
    const int bh = blockIdx.x; const int h = bh % H; const int b = bh / H;
    const int tid = threadIdx.x;
    const int i = tid & 63; const int seg = tid >> 6; const int j0 = seg * 16;
    float S[16];
    #pragma unroll
    for (int q = 0; q < 16; ++q) S[q] = 0.f;

    for (int t = 0; t < T; ++t) {
        size_t base = (size_t)(b * T + t) * C + h * 64;
        for (int u = tid; u < 384; u += 256) {
            int arr = u >> 6; int j = u & 63;
            float val;
            if      (arr == 0) val = r  [base + j];
            else if (arr == 1) val = wd [base + j];
            else if (arr == 2) val = k  [base + j];
            else if (arr == 3) val = v  [base + j];
            else if (arr == 4) val = -kkn[base + j];   // a_t = -kk
            else               val = bb [base + j];    // b_t = kk*a
            vec[u] = val;
        }
        __syncthreads();
        float sap = 0.f;
        #pragma unroll
        for (int q = 0; q < 16; ++q) sap += S[q] * vec[256 + j0 + q];
        psum[tid] = sap; __syncthreads();
        float sa = psum[i] + psum[64 + i] + psum[128 + i] + psum[192 + i];
        float vi = vec[192 + i];
        float yp = 0.f;
        #pragma unroll
        for (int q = 0; q < 16; ++q) {
            int j = j0 + q;
            float sv = S[q] * vec[64 + j] + sa * vec[320 + j] + vi * vec[128 + j];
            S[q] = sv;
            yp += sv * vec[j];
        }
        __syncthreads();            // all sa reads done before psum reuse
        psum[tid] = yp; __syncthreads();
        if (seg == 0)
            y[base + i] = psum[i] + psum[64 + i] + psum[128 + i] + psum[192 + i];
        // next iteration's barrier after vec-load orders psum reuse
    }
}

// ---------------------------------------------------------------------------
// GroupNorm(heads) + bonus + gate -> bf16   (block = 64, one per (m,h))
// ---------------------------------------------------------------------------
__global__ void __launch_bounds__(64)
k_gngate(const float* __restrict__ y, const float* __restrict__ r,
         const float* __restrict__ k, const float* __restrict__ v,
         const float* __restrict__ g, const float* __restrict__ lnw,
         const float* __restrict__ lnb, const float* __restrict__ rk,
         unsigned short* __restrict__ out, int H, int C)
{
    __shared__ float s[64];
    int bx = blockIdx.x; int h = bx % H; int m = bx / H;
    int tid = threadIdx.x;
    int c = h * 64 + tid;
    size_t idx = (size_t)m * C + c;
    float yv = y[idx];
    s[tid] = yv; __syncthreads();
    for (int o = 32; o > 0; o >>= 1) { if (tid < o) s[tid] += s[tid + o]; __syncthreads(); }
    float mean = s[0] * (1.0f / 64.0f); __syncthreads();
    float d = yv - mean;
    s[tid] = d * d; __syncthreads();
    for (int o = 32; o > 0; o >>= 1) { if (tid < o) s[tid] += s[tid + o]; __syncthreads(); }
    float rstd = rsqrtf(s[0] * (1.0f / 64.0f) + 64e-5f); __syncthreads();
    float gn = d * rstd * lnw[c] + lnb[c];
    s[tid] = r[idx] * k[idx] * rk[c]; __syncthreads();
    for (int o = 32; o > 0; o >>= 1) { if (tid < o) s[tid] += s[tid + o]; __syncthreads(); }
    float bonus = s[0];
    out[idx] = f2bf((gn + bonus * v[idx]) * g[idx]);
}

// ---------------------------------------------------------------------------
// Host driver
// ---------------------------------------------------------------------------
extern "C" void kernel_launch(void* const* d_in, const int* in_sizes, int n_in,
                              void* d_out, int out_size, void* d_ws, size_t ws_size,
                              hipStream_t stream)
{
    (void)in_sizes; (void)n_in; (void)out_size; (void)ws_size;
    const int L = 6, Bn = 2, T = 512, C = 768, H = 12, F = 3072;
    const int M = Bn * T;                 // 1024 token rows
    const size_t MC = (size_t)M * C;      // 786432
    const int Dw = 64, Da = 64, Dv = 32, Dg = 128;

    char* wsb = (char*)d_ws; size_t off = 0;
    auto alloc = [&](size_t bytes) -> char* {
        off = (off + 255) & ~(size_t)255;
        char* p = wsb + off; off += bytes; return p;
    };

    // bf16 transposed weights
    unsigned short* WrT  = (unsigned short*)alloc((size_t)L * C * C * 2);
    unsigned short* WkT  = (unsigned short*)alloc((size_t)L * C * C * 2);
    unsigned short* WvT  = (unsigned short*)alloc((size_t)L * C * C * 2);
    unsigned short* WoT  = (unsigned short*)alloc((size_t)L * C * C * 2);
    unsigned short* w1T  = (unsigned short*)alloc((size_t)L * C * Dw * 2);
    unsigned short* w2T  = (unsigned short*)alloc((size_t)L * Dw * C * 2);
    unsigned short* a1T  = (unsigned short*)alloc((size_t)L * C * Da * 2);
    unsigned short* a2T  = (unsigned short*)alloc((size_t)L * Da * C * 2);
    unsigned short* v1T  = (unsigned short*)alloc((size_t)L * C * Dv * 2);
    unsigned short* v2T  = (unsigned short*)alloc((size_t)L * Dv * C * 2);
    unsigned short* g1T  = (unsigned short*)alloc((size_t)L * C * Dg * 2);
    unsigned short* g2T  = (unsigned short*)alloc((size_t)L * Dg * C * 2);
    unsigned short* WckT = (unsigned short*)alloc((size_t)L * C * F * 2);
    unsigned short* WcvT = (unsigned short*)alloc((size_t)L * F * C * 2);

    // fp32 activations
    float* X      = (float*)alloc(MC * 4);
    float* LN     = (float*)alloc(MC * 4);
    float* Rf     = (float*)alloc(MC * 4);
    float* Kf     = (float*)alloc(MC * 4);
    float* Vf     = (float*)alloc(MC * 4);
    float* Af     = (float*)alloc(MC * 4);
    float* Dec    = (float*)alloc(MC * 4);
    float* Gf     = (float*)alloc(MC * 4);
    float* Yf     = (float*)alloc(MC * 4);
    float* Vfirst = (float*)alloc(MC * 4);
    float* KKn    = (float*)alloc(MC * 4);
    float* Bb     = (float*)alloc(MC * 4);
    float* TMPC   = (float*)alloc(MC * 4);
    float* TMPS   = (float*)alloc((size_t)M * 128 * 4);
    float* HF     = (float*)alloc((size_t)M * F * 4);

    // bf16 activations
    unsigned short* MIXR = (unsigned short*)alloc(MC * 2);
    unsigned short* MIXW = (unsigned short*)alloc(MC * 2);
    unsigned short* MIXK = (unsigned short*)alloc(MC * 2);
    unsigned short* MIXV = (unsigned short*)alloc(MC * 2);
    unsigned short* MIXA = (unsigned short*)alloc(MC * 2);
    unsigned short* MIXG = (unsigned short*)alloc(MC * 2);
    unsigned short* SMB  = (unsigned short*)alloc((size_t)M * 128 * 2);
    unsigned short* YBF  = (unsigned short*)alloc(MC * 2);
    unsigned short* CKB  = (unsigned short*)alloc(MC * 2);
    unsigned short* HFB  = (unsigned short*)alloc((size_t)M * F * 2);

    auto GB = [](long n) { return (unsigned)((n + 255) / 256); };
    auto tr = [&](int slot, unsigned short* dst, int K, int N) {
        long tot = (long)L * K * N;
        k_transpose_bf16<<<GB(tot), 256, 0, stream>>>((const float*)d_in[slot], dst, K, N, tot);
    };
    // weight conversions (all layers at once)
    tr(21, WrT, C, C);   tr(22, WkT, C, C);   tr(23, WvT, C, C);   tr(24, WoT, C, C);
    tr(8,  w1T, C, Dw);  tr(9,  w2T, Dw, C);
    tr(11, a1T, C, Da);  tr(12, a2T, Da, C);
    tr(14, v1T, C, Dv);  tr(15, v2T, Dv, C);
    tr(16, g1T, C, Dg);  tr(17, g2T, Dg, C);
    tr(32, WckT, C, F);  tr(33, WcvT, F, C);

    hipMemcpyAsync(X, d_in[0], MC * 4, hipMemcpyDeviceToDevice, stream);

    auto gemm = [&](const unsigned short* A, const unsigned short* Bt, float* Cc,
                    int Mm, int Nn, int Kk) {
        if (Nn % 64 == 0) {
            dim3 grid(Nn / 64, Mm / 128);
            gemm_bf16_wmma64<<<grid, 256, 0, stream>>>(A, Bt, Cc, Mm, Nn, Kk);
        } else {
            dim3 grid(Nn / 32, Mm / 128);
            gemm_bf16_wmma32<<<grid, 256, 0, stream>>>(A, Bt, Cc, Mm, Nn, Kk);
        }
    };

    for (int l = 0; l < L; ++l) {
        const float* xr  = (const float*)d_in[1]  + (size_t)l * C;
        const float* xw  = (const float*)d_in[2]  + (size_t)l * C;
        const float* xk  = (const float*)d_in[3]  + (size_t)l * C;
        const float* xv  = (const float*)d_in[4]  + (size_t)l * C;
        const float* xa  = (const float*)d_in[5]  + (size_t)l * C;
        const float* xg  = (const float*)d_in[6]  + (size_t)l * C;
        const float* w0  = (const float*)d_in[7]  + (size_t)l * C;
        const float* a0  = (const float*)d_in[10] + (size_t)l * C;
        const float* v0  = (const float*)d_in[13] + (size_t)l * C;
        const float* kkv = (const float*)d_in[18] + (size_t)l * C;
        const float* kav = (const float*)d_in[19] + (size_t)l * C;
        const float* rkp = (const float*)d_in[20] + (size_t)l * C;   // [H,64] == C
        const float* lnxw = (const float*)d_in[25] + (size_t)l * C;
        const float* lnxb = (const float*)d_in[26] + (size_t)l * C;
        const float* ln1w = (const float*)d_in[27] + (size_t)l * C;
        const float* ln1b = (const float*)d_in[28] + (size_t)l * C;
        const float* ln2w = (const float*)d_in[29] + (size_t)l * C;
        const float* ln2b = (const float*)d_in[30] + (size_t)l * C;
        const float* cxk  = (const float*)d_in[31] + (size_t)l * C;

        const unsigned short* WrTl  = WrT  + (size_t)l * C * C;
        const unsigned short* WkTl  = WkT  + (size_t)l * C * C;
        const unsigned short* WvTl  = WvT  + (size_t)l * C * C;
        const unsigned short* WoTl  = WoT  + (size_t)l * C * C;
        const unsigned short* w1Tl  = w1T  + (size_t)l * C * Dw;
        const unsigned short* w2Tl  = w2T  + (size_t)l * Dw * C;
        const unsigned short* a1Tl  = a1T  + (size_t)l * C * Da;
        const unsigned short* a2Tl  = a2T  + (size_t)l * Da * C;
        const unsigned short* v1Tl  = v1T  + (size_t)l * C * Dv;
        const unsigned short* v2Tl  = v2T  + (size_t)l * Dv * C;
        const unsigned short* g1Tl  = g1T  + (size_t)l * C * Dg;
        const unsigned short* g2Tl  = g2T  + (size_t)l * Dg * C;
        const unsigned short* WckTl = WckT + (size_t)l * C * F;
        const unsigned short* WcvTl = WcvT + (size_t)l * F * C;

        // ---- time mix ----
        k_layernorm<<<M, 256, 0, stream>>>(X, ln1w, ln1b, LN, C);
        k_mix6<<<GB(MC), 256, 0, stream>>>(LN, xr, xw, xk, xv, xa, xg,
                                           MIXR, MIXW, MIXK, MIXV, MIXA, MIXG,
                                           T, C, (int)MC);
        gemm(MIXR, WrTl, Rf, M, C, C);
        gemm(MIXK, WkTl, Kf, M, C, C);
        gemm(MIXV, WvTl, Vf, M, C, C);

        if (l == 0) {
            hipMemcpyAsync(Vfirst, Vf, MC * 4, hipMemcpyDeviceToDevice, stream);
        } else {
            gemm(MIXV, v1Tl, TMPS, M, Dv, C);
            k_convert<<<GB((long)M * Dv), 256, 0, stream>>>(TMPS, SMB, M * Dv, 0);
            gemm(SMB, v2Tl, TMPC, M, C, Dv);
            k_vmix<<<GB(MC), 256, 0, stream>>>(Vf, Vfirst, TMPC, v0, C, (int)MC);
        }
        // w chain -> decay
        gemm(MIXW, w1Tl, TMPS, M, Dw, C);
        k_convert<<<GB((long)M * Dw), 256, 0, stream>>>(TMPS, SMB, M * Dw, 1);
        gemm(SMB, w2Tl, TMPC, M, C, Dw);
        k_decay<<<GB(MC), 256, 0, stream>>>(TMPC, w0, Dec, C, (int)MC);
        // a chain
        gemm(MIXA, a1Tl, TMPS, M, Da, C);
        k_convert<<<GB((long)M * Da), 256, 0, stream>>>(TMPS, SMB, M * Da, 0);
        gemm(SMB, a2Tl, TMPC, M, C, Da);
        k_asig<<<GB(MC), 256, 0, stream>>>(TMPC, a0, Af, C, (int)MC);
        // g chain
        gemm(MIXG, g1Tl, TMPS, M, Dg, C);
        k_convert<<<GB((long)M * Dg), 256, 0, stream>>>(TMPS, SMB, M * Dg, 2);
        gemm(SMB, g2Tl, Gf, M, C, Dg);

        k_kkprep<<<M * H, 64, 0, stream>>>(Kf, Af, kkv, kav, KKn, Bb, H, C);
        k_wkv<<<Bn * H, 256, 0, stream>>>(Rf, Dec, Kf, Vf, KKn, Bb, Yf, T, C, H);
        k_gngate<<<M * H, 64, 0, stream>>>(Yf, Rf, Kf, Vf, Gf, lnxw, lnxb, rkp, YBF, H, C);

        gemm(YBF, WoTl, TMPC, M, C, C);
        k_add<<<GB(MC), 256, 0, stream>>>(X, TMPC, (int)MC);

        // ---- channel mix ----
        k_layernorm<<<M, 256, 0, stream>>>(X, ln2w, ln2b, LN, C);
        k_cmixin<<<GB(MC), 256, 0, stream>>>(LN, cxk, CKB, T, C, (int)MC);
        gemm(CKB, WckTl, HF, M, F, C);
        k_convert<<<GB((long)M * F), 256, 0, stream>>>(HF, HFB, M * F, 3);
        gemm(HFB, WcvTl, TMPC, M, C, F);
        k_add<<<GB(MC), 256, 0, stream>>>(X, TMPC, (int)MC);
    }

    hipMemcpyAsync(d_out, X, MC * 4, hipMemcpyDeviceToDevice, stream);
}
